// MultiHeadAttention_17978733101316
// MI455X (gfx1250) — matrix operand order repaired and checked by reading, not compiled, not run on hardware
//
#include <hip/hip_runtime.h>
#include <hip/hip_bf16.h>

// ---------------------------------------------------------------------------
// MHA forward for MI455X (gfx1250, wave32, WMMA, 320KB LDS/WGP).
// f32->bf16 convert -> 3 projection WMMA GEMMs (16x64 strip per wave) ->
// fused attention: one block per (b, 16-query tile), loops over all 16 heads,
// accumulates avg_attn in LDS (no global atomics) -> output WMMA GEMM (f32).
// ---------------------------------------------------------------------------

#define BB 2
#define SS 2048
#define EE_DIM 1024
#define HH 16
#define DD 64

typedef __attribute__((ext_vector_type(16))) __bf16 v16bf;
typedef __attribute__((ext_vector_type(8)))  float  v8f;

union Frag {
    v16bf bf;
    unsigned int u32[8];
    unsigned short u16[16];
};

static __device__ __forceinline__ unsigned short f32_to_bf16(float f) {
    unsigned int u = __float_as_uint(f);
    u = u + 0x7FFFu + ((u >> 16) & 1u);   // round-to-nearest-even
    return (unsigned short)(u >> 16);
}

// --------------------------- f32 -> bf16 convert ---------------------------
__global__ void cvt_bf16_kernel(const float* __restrict__ src,
                                unsigned short* __restrict__ dst, int n) {
    int i = blockIdx.x * blockDim.x + threadIdx.x;
    if (i < n) dst[i] = f32_to_bf16(src[i]);
}

// ---------------------------------------------------------------------------
// GEMM:  Y[m,n] = sum_k X[m,k] * W[n,k] + bias[n]     (i.e. X @ W^T + b)
// X: bf16 [Mdim,Kdim] row-major.  W: bf16 [Ndim,Kdim] row-major ([out,in]).
// One 16x64 output strip per wave (A fragment reused across 4 WMMAs);
// 8 waves per 256-thread block.
// MODE 0: store bf16 to [B,H,S,D]   (m->(b,s), n->(h,d))        (Q, K)
// MODE 1: store bf16 to [B,H,D,S]   (V transposed for attn@V)
// MODE 2: store f32  to [m*Ndim+n]  (final output)
// ---------------------------------------------------------------------------
template <int MODE>
__global__ void gemm_xwT_kernel(const unsigned short* __restrict__ X,
                                const unsigned short* __restrict__ W,
                                const float* __restrict__ bias,
                                void* __restrict__ out,
                                int Mdim, int Ndim, int Kdim) {
    const int wid  = blockIdx.x * (blockDim.x >> 5) + (threadIdx.x >> 5);
    const int lane = threadIdx.x & 31;
    const int stripsN = Ndim >> 6;                 // 64-wide N strips
    const int tm = wid / stripsN;
    const int sn = wid % stripsN;
    if (tm >= (Mdim >> 4)) return;                 // wave-uniform
    const int m0 = tm << 4, n0 = sn << 6;
    const int l15  = lane & 15;
    const int hi8  = (lane & 16) >> 1;             // A / C lane-half shift
    const int hi16 = (lane & 16);                  // unused (B also shifts K by hi8)

    const unsigned short* Arow = X + (size_t)(m0 + l15) * Kdim;
    const unsigned short* Brow0 = W + (size_t)(n0 + l15) * Kdim;

    v8f acc0 = {}, acc1 = {}, acc2 = {}, acc3 = {};
    for (int k0 = 0; k0 < Kdim; k0 += 32) {
        Frag a;
#pragma unroll
        for (int j = 0; j < 8; ++j)
            a.u32[j] = *(const unsigned int*)(Arow + k0 + 2 * j +
                                              ((j >= 4) ? 8 : 0) + hi8);
        __builtin_prefetch(Arow + k0 + 256, 0, 1);   // global_prefetch_b8
#pragma unroll
        for (int t = 0; t < 4; ++t) {
            Frag b;
            const unsigned short* Brow = Brow0 + (size_t)(t * 16) * Kdim;
#pragma unroll
            for (int j = 0; j < 8; ++j)
                b.u32[j] = *(const unsigned int*)(Brow + k0 + 2 * j + ((j >= 4) ? 8 : 0) + hi8);
            v8f* accp = (t == 0) ? &acc0 : (t == 1) ? &acc1
                       : (t == 2) ? &acc2 : &acc3;
            *accp = __builtin_amdgcn_wmma_f32_16x16x32_bf16(
                false, a.bf, false, b.bf, (short)0, *accp, false, false);
        }
    }

#pragma unroll
    for (int t = 0; t < 4; ++t) {
        const v8f acc = (t == 0) ? acc0 : (t == 1) ? acc1
                       : (t == 2) ? acc2 : acc3;
#pragma unroll
        for (int j = 0; j < 8; ++j) {
            const int m = m0 + j + hi8;
            const int n = n0 + t * 16 + l15;
            const float v = acc[j] + bias[n];
            if constexpr (MODE == 2) {
                ((float*)out)[(size_t)m * Ndim + n] = v;
            } else {
                const int b_ = m >> 11, s_ = m & (SS - 1);   // m = b*S + s
                const int h_ = n >> 6,  d_ = n & (DD - 1);   // n = h*D + d
                size_t idx;
                if constexpr (MODE == 0)
                    idx = (((size_t)b_ * HH + h_) * SS + s_) * DD + d_;
                else
                    idx = (((size_t)b_ * HH + h_) * DD + d_) * SS + s_;
                ((unsigned short*)out)[idx] = f32_to_bf16(v);
            }
        }
    }
}

// ---------------------------------------------------------------------------
// Fused attention: one block per (b, 16-query tile); loops over all 16 heads.
// LDS: sc[16][2048] scores (128KB) + av[16][2048] head-average accumulator
// (128KB) + reduce/partial scratch -> ~261KB, fits CDNA5's 320KB workgroup
// LDS.  avg_attn is written with plain stores -- no global atomics.
// 256 threads = 8 waves.
// ---------------------------------------------------------------------------
__global__ void attn_kernel(const unsigned short* __restrict__ Qh,
                            const unsigned short* __restrict__ Kh,
                            const unsigned short* __restrict__ Vt,
                            const int* __restrict__ mask,
                            unsigned short* __restrict__ AO,
                            float* __restrict__ avg) {
    extern __shared__ float smem[];
    float* sc   = smem;                   // [16][2048] scores / probs
    float* av   = sc + 16 * SS;           // [16][2048] sum over heads / H
    float* red  = av + 16 * SS;           // [256] softmax reduce
    float* pacc = red + 256;              // [4][16][16] attn@V partial sums

    const int qt = blockIdx.x & ((SS / 16) - 1);
    const int b_ = blockIdx.x >> 7;
    const int q0 = qt << 4;
    const int tid  = threadIdx.x;         // 0..255
    const int wave = tid >> 5;            // 0..7
    const int lane = tid & 31;
    const int l15  = lane & 15;
    const int hi8  = (lane & 16) >> 1;
    const int hi16 = (lane & 16);

    // zero the head-average accumulator
    for (int i = tid; i < 16 * SS; i += 256) av[i] = 0.0f;

    for (int h_ = 0; h_ < HH; ++h_) {
        const size_t headRow = ((size_t)b_ * HH + h_) * SS;

        // hoist the Q-tile A-fragments (invariant over the 128 key tiles)
        const unsigned short* Qbase = Qh + (headRow + q0 + l15) * DD;
        Frag aq[2];
#pragma unroll
        for (int kk2 = 0; kk2 < 2; ++kk2)
#pragma unroll
            for (int j = 0; j < 8; ++j)
                aq[kk2].u32[j] = *(const unsigned int*)(
                    Qbase + kk2 * 32 + 2 * j + ((j >= 4) ? 8 : 0) + hi8);

        __syncthreads();   // prev head done reading sc / pacc

        // ---- scores = (Q K^T)/sqrt(D), masked, into LDS ----
        for (int kt = wave; kt < SS / 16; kt += 8) {
            const int k0 = kt << 4;
            const unsigned short* Kbase = Kh + (headRow + k0 + l15) * DD;
            v8f acc = {};
#pragma unroll
            for (int kk2 = 0; kk2 < 2; ++kk2) {
                Frag b;
#pragma unroll
                for (int j = 0; j < 8; ++j)
                    b.u32[j] = *(const unsigned int*)(
                        Kbase + kk2 * 32 + 2 * j + ((j >= 4) ? 8 : 0) + hi8);
                acc = __builtin_amdgcn_wmma_f32_16x16x32_bf16(
                    false, aq[kk2].bf, false, b.bf, (short)0, acc, false, false);
            }
#pragma unroll
            for (int j = 0; j < 8; ++j) {
                const int qr = j + hi8;
                const int kc = k0 + l15;
                const int mv = mask[((size_t)b_ * SS + q0 + qr) * SS + kc];
                sc[qr * SS + kc] = (mv == 0) ? -1e9f : acc[j] * 0.125f;
            }
        }
        __syncthreads();

        // ---- row softmax: 16 rows x 16 threads ----
        const int r = tid >> 4, cid = tid & 15;
        float mx = -3.4e38f;
        for (int c = cid; c < SS; c += 16) mx = fmaxf(mx, sc[r * SS + c]);
        red[tid] = mx;
        __syncthreads();
        if (cid == 0) {
            float m = red[r * 16];
            for (int i = 1; i < 16; ++i) m = fmaxf(m, red[r * 16 + i]);
            red[r * 16] = m;
        }
        __syncthreads();
        mx = red[r * 16];
        __syncthreads();
        float sum = 0.0f;
        for (int c = cid; c < SS; c += 16) {
            const float e = __expf(sc[r * SS + c] - mx);
            sc[r * SS + c] = e;
            sum += e;
        }
        red[tid] = sum;
        __syncthreads();
        if (cid == 0) {
            float s2 = red[r * 16];
            for (int i = 1; i < 16; ++i) s2 += red[r * 16 + i];
            red[r * 16] = s2;
        }
        __syncthreads();
        const float inv = 1.0f / red[r * 16];
        for (int c = cid; c < SS; c += 16) {
            const float p = sc[r * SS + c] * inv;
            sc[r * SS + c] = p;
            av[r * SS + c] += p * (1.0f / (float)HH);   // LDS, race-free
        }
        __syncthreads();

        // ---- out = attn @ V ----
        // wave w: d-columns [(w&3)*16, +16), K half (w>>2)*1024.
        const int d0 = (wave & 3) << 4;
        const int kbeg = (wave >> 2) << 10;
        const unsigned short* Vbase =
            Vt + (((size_t)b_ * HH + h_) * DD + d0 + l15) * SS;
        v8f acc = {};
        for (int k0 = kbeg; k0 < kbeg + 1024; k0 += 32) {
            Frag a, b;
#pragma unroll
            for (int j = 0; j < 8; ++j) {
                const int ka = k0 + 2 * j + ((j >= 4) ? 8 : 0) + hi8;
                const float f0 = sc[l15 * SS + ka];
                const float f1 = sc[l15 * SS + ka + 1];
                a.u32[j] = (unsigned int)f32_to_bf16(f0) |
                           ((unsigned int)f32_to_bf16(f1) << 16);
                b.u32[j] = *(const unsigned int*)(Vbase + k0 + 2 * j + ((j >= 4) ? 8 : 0) + hi8);
            }
            acc = __builtin_amdgcn_wmma_f32_16x16x32_bf16(
                false, a.bf, false, b.bf, (short)0, acc, false, false);
        }
        if (wave >= 4) {   // stash upper-K partials
#pragma unroll
            for (int j = 0; j < 8; ++j)
                pacc[(wave & 3) * 256 + (j + hi8) * 16 + l15] = acc[j];
        }
        __syncthreads();
        if (wave < 4) {    // combine halves, emit bf16 output tile
#pragma unroll
            for (int j = 0; j < 8; ++j) {
                const int qr = j + hi8;
                const int dc = d0 + l15;
                const float v = acc[j] + pacc[(wave & 3) * 256 + qr * 16 + l15];
                AO[((size_t)b_ * SS + q0 + qr) * EE_DIM + h_ * DD + dc] =
                    f32_to_bf16(v);
            }
        }
    }

    // ---- write avg_attn rows with plain stores ----
    __syncthreads();
    for (int i = tid; i < 16 * SS; i += 256) {
        const int r = i >> 11;
        const int c = i & (SS - 1);
        avg[((size_t)b_ * SS + q0 + r) * SS + c] = av[i];
    }
}

// ---------------------------------------------------------------------------
extern "C" void kernel_launch(void* const* d_in, const int* in_sizes, int n_in,
                              void* d_out, int out_size, void* d_ws,
                              size_t ws_size, hipStream_t stream) {
    (void)in_sizes; (void)n_in; (void)out_size; (void)ws_size;

    const float* q    = (const float*)d_in[0];
    const float* k    = (const float*)d_in[1];
    const float* v    = (const float*)d_in[2];
    const int*   mask = (const int*)  d_in[3];
    const float* Wq   = (const float*)d_in[4];
    const float* bq   = (const float*)d_in[5];
    const float* Wk   = (const float*)d_in[6];
    const float* bk   = (const float*)d_in[7];
    const float* Wv   = (const float*)d_in[8];
    const float* bv   = (const float*)d_in[9];
    const float* Wo   = (const float*)d_in[10];
    const float* bo   = (const float*)d_in[11];

    float* out = (float*)d_out;                       // [B,S,E]
    float* avg = out + (size_t)BB * SS * EE_DIM;      // [B,S,S]

    const int nBSE = BB * SS * EE_DIM;                // 4,194,304
    const int nEE  = EE_DIM * EE_DIM;                 // 1,048,576

    // workspace layout (all bf16 as ushort)
    unsigned short* w = (unsigned short*)d_ws;
    unsigned short* Xq  = w;              w += nBSE;
    unsigned short* Xk  = w;              w += nBSE;
    unsigned short* Xv  = w;              w += nBSE;
    unsigned short* Wqb = w;              w += nEE;
    unsigned short* Wkb = w;              w += nEE;
    unsigned short* Wvb = w;              w += nEE;
    unsigned short* Wob = w;              w += nEE;
    unsigned short* Qh  = w;              w += nBSE;  // [B,H,S,D]
    unsigned short* Kh  = w;              w += nBSE;  // [B,H,S,D]
    unsigned short* Vt  = w;              w += nBSE;  // [B,H,D,S]
    unsigned short* AO  = w;              w += nBSE;  // [B,S,E]

    // 1) convert inputs + weights to bf16
    cvt_bf16_kernel<<<nBSE / 256, 256, 0, stream>>>(q,  Xq,  nBSE);
    cvt_bf16_kernel<<<nBSE / 256, 256, 0, stream>>>(k,  Xk,  nBSE);
    cvt_bf16_kernel<<<nBSE / 256, 256, 0, stream>>>(v,  Xv,  nBSE);
    cvt_bf16_kernel<<<nEE  / 256, 256, 0, stream>>>(Wq, Wqb, nEE);
    cvt_bf16_kernel<<<nEE  / 256, 256, 0, stream>>>(Wk, Wkb, nEE);
    cvt_bf16_kernel<<<nEE  / 256, 256, 0, stream>>>(Wv, Wvb, nEE);
    cvt_bf16_kernel<<<nEE  / 256, 256, 0, stream>>>(Wo, Wob, nEE);

    // 2) projections: (4096/16)*(1024/64) = 4096 waves / 8 = 512 blocks
    const int Mdim = BB * SS;
    const int gemmBlocks = (Mdim / 16) * (EE_DIM / 64) / 8;
    gemm_xwT_kernel<0><<<gemmBlocks, 256, 0, stream>>>(Xq, Wqb, bq, Qh,
                                                       Mdim, EE_DIM, EE_DIM);
    gemm_xwT_kernel<0><<<gemmBlocks, 256, 0, stream>>>(Xk, Wkb, bk, Kh,
                                                       Mdim, EE_DIM, EE_DIM);
    gemm_xwT_kernel<1><<<gemmBlocks, 256, 0, stream>>>(Xv, Wvb, bv, Vt,
                                                       Mdim, EE_DIM, EE_DIM);

    // 3) fused attention: one block per (b, q-tile); ~261KB dynamic LDS
    const size_t shmem =
        (size_t)(2 * 16 * SS + 256 + 4 * 256) * sizeof(float);
    attn_kernel<<<BB * (SS / 16), 256, shmem, stream>>>(Qh, Kh, Vt, mask,
                                                        AO, avg);

    // 4) output projection (f32 result straight to d_out)
    gemm_xwT_kernel<2><<<gemmBlocks, 256, 0, stream>>>(AO, Wob, bo, out,
                                                       Mdim, EE_DIM, EE_DIM);
}
